// S2S_BA_Model_54296976556163
// MI455X (gfx1250) — compile-verified, matrix operand order checked by reading, not run on hardware
//
#include <hip/hip_runtime.h>
#include <hip/hip_bf16.h>

// Problem constants (from reference)
#define BSZ 512
#define TSEQ 336
#define FIN 64
#define HID 256
#define TL 48
#define G3 (3 * HID)   // 768
#define SPAD 8         // LDS row padding (halfs) to spread rows across banks

typedef __attribute__((ext_vector_type(16))) _Float16 v16h;
typedef __attribute__((ext_vector_type(8)))  float    v8f;

// ---------------------------------------------------------------------------
// WMMA helpers (wave32, V_WMMA_F32_16X16X32_F16)
//
// A (16x32 f16): lane L -> row M = L&15, half = L>>4.
//   VGPR r (0..7): K pair base = (r>>2)*16 + half*8 + (r&3)*2
// B (32x16 f16): lane L -> col N = L&15, K packing mirrors A.
// C/D (16x16 f32): lane L -> col N = L&15; VGPR r -> row M = r + 8*half.
// ---------------------------------------------------------------------------

__device__ __forceinline__ v8f wmma_f16(v16h a, v16h b, v8f c) {
  return __builtin_amdgcn_wmma_f32_16x16x32_f16(false, a, false, b,
                                                (short)0, c, false, false);
}

// Load a 16x32 A-fragment from an LDS-staged f16 tile (row stride `ld` halfs).
__device__ __forceinline__ v16h load_a_lds(const _Float16* S, int ld,
                                           int kbase, int lane) {
  union { v16h v; unsigned int u[8]; } u;
  const int m = lane & 15, half = lane >> 4;
  const _Float16* rp = S + m * ld + kbase;
#pragma unroll
  for (int r = 0; r < 8; ++r) {
    const int k = ((r >> 2) << 4) + (half << 3) + ((r & 3) << 1);
    u.u[r] = *(const unsigned int*)(rp + k);   // ds_load, 4B aligned
  }
  return u.v;
}

// Load a 32x16 B-tile from an f16 row-major weight matrix W[N_total][K_total]:
// B[k][n] = W[gbase + n][kbase + k]  (i.e. computing  A * W^T).
__device__ __forceinline__ v16h load_b_f16(const _Float16* __restrict__ W,
                                           int K, int gbase, int kbase,
                                           int lane) {
  union { v16h v; unsigned int u[8]; } u;
  const int n = lane & 15, half = lane >> 4;
  const _Float16* rp = W + (size_t)(gbase + n) * K + kbase;
#pragma unroll
  for (int r = 0; r < 8; ++r) {
    const int k = ((r >> 2) << 4) + (half << 3) + ((r & 3) << 1);
    u.u[r] = *(const unsigned int*)(rp + k);   // merges into global_load_b128
  }
  return u.v;
}

// Cooperative block stage: LDS[m][k] (padded) = (f16)src[(rowbase+m)*ld + k]
__device__ __forceinline__ void stage_a(const float* __restrict__ src, int ld,
                                        int rowbase, int kA, _Float16* dst) {
  const int n = 16 * kA;
  for (int i = threadIdx.x; i < n; i += 256) {
    const int m = i / kA, k = i % kA;
    dst[m * (kA + SPAD) + k] = (_Float16)src[(size_t)(rowbase + m) * ld + k];
  }
}

__device__ __forceinline__ float wave_sum(float v) {
#pragma unroll
  for (int o = 16; o > 0; o >>= 1) v += __shfl_xor(v, o, 32);
  return v;
}
__device__ __forceinline__ float wave_max(float v) {
#pragma unroll
  for (int o = 16; o > 0; o >>= 1) v = fmaxf(v, __shfl_xor(v, o, 32));
  return v;
}
__device__ __forceinline__ float sigmoidf(float x) {
  return 1.0f / (1.0f + expf(-x));
}

// ---------------------------------------------------------------------------
// Fused GRU step (encoder or decoder). One wave per 16x16 tile of h.
// 64 blocks x 256 threads; block-uniform row tile (rowbase = blockIdx>>1),
// wave-private col tile. A tiles (x rows and h rows) staged once in LDS.
// ---------------------------------------------------------------------------
__global__ __launch_bounds__(256) void k_gru_step(
    const float* __restrict__ xA, int ldxA, int kA,
    const _Float16* __restrict__ wih,
    const float* __restrict__ hA,
    const _Float16* __restrict__ whh,
    const float* __restrict__ bih, const float* __restrict__ bhh,
    const float* __restrict__ r1v, const float* __restrict__ r1w,
    float* __restrict__ h_out, float* __restrict__ enc_out) {
  __shared__ _Float16 sX[16 * (HID + SPAD)];
  __shared__ _Float16 sH[16 * (HID + SPAD)];
  const int lane = threadIdx.x & 31;
  const int rowbase = (blockIdx.x >> 1) << 4;                     // 0..496
  const int colbase = (((threadIdx.x >> 5) + ((blockIdx.x & 1) << 3))) << 4;

  stage_a(xA, ldxA, rowbase, kA, sX);
  stage_a(hA, HID, rowbase, HID, sH);
  __syncthreads();

  v8f ar = {}, az = {}, ain = {}, ahn = {};

  // gi: x-side GEMM (r,z fold into shared accumulators; n kept separate)
  for (int kb = 0; kb < kA; kb += 32) {
    v16h a = load_a_lds(sX, kA + SPAD, kb, lane);
    ar  = wmma_f16(a, load_b_f16(wih, kA, colbase,           kb, lane), ar);
    az  = wmma_f16(a, load_b_f16(wih, kA, HID + colbase,     kb, lane), az);
    ain = wmma_f16(a, load_b_f16(wih, kA, 2 * HID + colbase, kb, lane), ain);
  }
  // gh: h-side GEMM
#pragma unroll 2
  for (int kb = 0; kb < HID; kb += 32) {
    v16h a = load_a_lds(sH, HID + SPAD, kb, lane);
    ar  = wmma_f16(a, load_b_f16(whh, HID, colbase,           kb, lane), ar);
    az  = wmma_f16(a, load_b_f16(whh, HID, HID + colbase,     kb, lane), az);
    ahn = wmma_f16(a, load_b_f16(whh, HID, 2 * HID + colbase, kb, lane), ahn);
  }

  const int half = lane >> 4;
  const int g = colbase + (lane & 15);
  const float b_r  = bih[g] + bhh[g];
  const float b_z  = bih[HID + g] + bhh[HID + g];
  const float b_in = bih[2 * HID + g];
  const float b_hn = bhh[2 * HID + g];

#pragma unroll
  for (int e = 0; e < 8; ++e) {
    const int row = rowbase + e + 8 * half;
    float gr = ar[e] + b_r, gz = az[e] + b_z;
    float gin = ain[e] + b_in, ghn = ahn[e] + b_hn;
    if (r1w) {
      const float p = r1v[row];
      gr  += p * r1w[g];
      gz  += p * r1w[HID + g];
      gin += p * r1w[2 * HID + g];
    }
    const float r = sigmoidf(gr);
    const float z = sigmoidf(gz);
    const float n = tanhf(gin + r * ghn);
    const float hold = hA[(size_t)row * HID + g];
    const float hnew = (1.0f - z) * n + z * hold;
    h_out[(size_t)row * HID + g] = hnew;
    if (enc_out) enc_out[(size_t)row * HID + g] = hnew;
  }
}

// ---------------------------------------------------------------------------
// GEMM: C[m][n] = sum_k A[m][k] * W[n][k] (+ bias[n]).  N fixed = 256.
// A: f32 row-major (lda), staged per block in LDS; W: f16 row-major [256][K].
// Launch 2 blocks per 16-row stripe (Mtiles*2 blocks of 256 threads).
// ---------------------------------------------------------------------------
__global__ __launch_bounds__(256) void k_gemm(
    const float* __restrict__ A, int lda,
    const _Float16* __restrict__ W, int K,
    const float* __restrict__ bias,
    float* __restrict__ C, int ldc) {
  __shared__ _Float16 sA[16 * (HID + SPAD)];
  const int lane = threadIdx.x & 31;
  const int rowbase = (blockIdx.x >> 1) << 4;
  const int colbase = (((threadIdx.x >> 5) + ((blockIdx.x & 1) << 3))) << 4;

  stage_a(A, lda, rowbase, K, sA);
  __syncthreads();

  v8f acc = {};
#pragma unroll 2
  for (int kb = 0; kb < K; kb += 32) {
    v16h a = load_a_lds(sA, K + SPAD, kb, lane);
    acc = wmma_f16(a, load_b_f16(W, K, colbase, kb, lane), acc);
  }
  const int half = lane >> 4;
  const int n = colbase + (lane & 15);
  const float bv = bias ? bias[n] : 0.0f;
#pragma unroll
  for (int e = 0; e < 8; ++e)
    C[(size_t)(rowbase + e + 8 * half) * ldc + n] = acc[e] + bv;
}

// ---------------------------------------------------------------------------
// Decoder output projection:
// out[b][g] = pred[b]*wp[g] + h[b]@Wh[g]^T + ctx[b]@Wc[g]^T + bout[g]
// wout: f16 [256][512], cols [0:256)=h part, [256:512)=ctx part.
// ---------------------------------------------------------------------------
__global__ __launch_bounds__(256) void k_dec_out(
    const float* __restrict__ h, const float* __restrict__ ctx,
    const _Float16* __restrict__ wout, const float* __restrict__ wp,
    const float* __restrict__ bout, const float* __restrict__ pred,
    float* __restrict__ outv) {
  __shared__ _Float16 sH[16 * (HID + SPAD)];
  __shared__ _Float16 sC[16 * (HID + SPAD)];
  const int lane = threadIdx.x & 31;
  const int rowbase = (blockIdx.x >> 1) << 4;
  const int colbase = (((threadIdx.x >> 5) + ((blockIdx.x & 1) << 3))) << 4;

  stage_a(h, HID, rowbase, HID, sH);
  stage_a(ctx, HID, rowbase, HID, sC);
  __syncthreads();

  v8f acc = {};
#pragma unroll 2
  for (int kb = 0; kb < HID; kb += 32) {
    v16h ah = load_a_lds(sH, HID + SPAD, kb, lane);
    acc = wmma_f16(ah, load_b_f16(wout, 2 * HID, colbase, kb, lane), acc);
    v16h ac = load_a_lds(sC, HID + SPAD, kb, lane);
    acc = wmma_f16(ac, load_b_f16(wout, 2 * HID, colbase, HID + kb, lane), acc);
  }
  const int half = lane >> 4;
  const int g = colbase + (lane & 15);
  const float wpg = wp[g], bg = bout[g];
#pragma unroll
  for (int e = 0; e < 8; ++e) {
    const int row = rowbase + e + 8 * half;
    outv[(size_t)row * HID + g] = acc[e] + pred[row] * wpg + bg;
  }
}

// ---------------------------------------------------------------------------
// score[t][b] = sum_h v[h] * tanh(hW1[b][h] + enc_proj[t][b][h])
// One wave per (t,b). enc_proj is streamed non-temporally (176MB/step, no
// reuse) so the 192MB L2 keeps encoder_outputs resident across all 48 steps.
// ---------------------------------------------------------------------------
__global__ __launch_bounds__(256) void k_score(
    const float* __restrict__ hW1, const float* __restrict__ eproj,
    const float* __restrict__ v, float* __restrict__ score) {
  const int lane = threadIdx.x & 31;
  const int wid  = blockIdx.x * 8 + (threadIdx.x >> 5);
  const int t = wid >> 9, b = wid & 511;
  const float* hp = hW1 + (size_t)b * HID;
  const float* ep = eproj + ((size_t)t * BSZ + b) * HID;
  float s = 0.0f;
#pragma unroll
  for (int h = lane; h < HID; h += 32)
    s += v[h] * tanhf(hp[h] + __builtin_nontemporal_load(ep + h));
  s = wave_sum(s);
  if (lane == 0) score[(size_t)t * BSZ + b] = s;
}

// softmax over t (axis 0). One wave per batch element b: 64 blocks.
__global__ __launch_bounds__(256) void k_softmax(
    const float* __restrict__ score, float* __restrict__ attn) {
  const int lane = threadIdx.x & 31;
  const int b = blockIdx.x * 8 + (threadIdx.x >> 5);
  float m = -3.4e38f;
  for (int t = lane; t < TSEQ; t += 32) m = fmaxf(m, score[(size_t)t * BSZ + b]);
  m = wave_max(m);
  float s = 0.0f;
  for (int t = lane; t < TSEQ; t += 32) {
    const float e = expf(score[(size_t)t * BSZ + b] - m);
    attn[(size_t)t * BSZ + b] = e;
    s += e;
  }
  s = wave_sum(s);
  const float inv = 1.0f / s;
  for (int t = lane; t < TSEQ; t += 32) attn[(size_t)t * BSZ + b] *= inv;
}

// context[b][h] = sum_t attn[t][b] * enc_out[t][b][h]. One block per b.
// Attention weights staged in LDS; enc_out reads stay regular-temporal so
// they remain L2-resident for the next decoder step.
__global__ __launch_bounds__(256) void k_context(
    const float* __restrict__ attn, const float* __restrict__ encout,
    float* __restrict__ ctx) {
  __shared__ float s_attn[TSEQ];
  const int b = blockIdx.x;
  for (int i = threadIdx.x; i < TSEQ; i += 256) s_attn[i] = attn[(size_t)i * BSZ + b];
  __syncthreads();
  const int h = threadIdx.x;
  float s = 0.0f;
  for (int t = 0; t < TSEQ; ++t)
    s += s_attn[t] * encout[((size_t)t * BSZ + b) * HID + h];
  ctx[(size_t)b * HID + h] = s;
}

// pred[b] = src[b] . wu + bu ; optionally write d_out column. Wave per b.
__global__ __launch_bounds__(256) void k_pred(
    const float* __restrict__ src, const float* __restrict__ wu,
    const float* __restrict__ bu, float* __restrict__ pred,
    float* __restrict__ outcol, int ldout) {
  const int lane = threadIdx.x & 31;
  const int b = blockIdx.x * 8 + (threadIdx.x >> 5);
  float s = 0.0f;
#pragma unroll
  for (int h = lane; h < HID; h += 32) s += src[(size_t)b * HID + h] * wu[h];
  s = wave_sum(s);
  if (lane == 0) {
    const float p = s + bu[0];
    pred[b] = p;
    if (outcol) outcol[(size_t)b * ldout] = p;
  }
}

// f32 -> f16 weight repack: dst[r*cols + c] = src[r*src_ld + src_off + c]
__global__ void k_cvt_f16(const float* __restrict__ src, _Float16* __restrict__ dst,
                          int rows, int src_ld, int src_off, int cols) {
  const int n = rows * cols;
  for (int i = blockIdx.x * blockDim.x + threadIdx.x; i < n;
       i += gridDim.x * blockDim.x) {
    const int r = i / cols, c = i % cols;
    dst[i] = (_Float16)src[(size_t)r * src_ld + src_off + c];
  }
}

// strided f32 gather: dst[i] = src[i*stride]
__global__ void k_gather_f32(const float* __restrict__ src, float* __restrict__ dst,
                             int n, int stride) {
  for (int i = blockIdx.x * blockDim.x + threadIdx.x; i < n;
       i += gridDim.x * blockDim.x)
    dst[i] = src[(size_t)i * stride];
}

// ---------------------------------------------------------------------------
extern "C" void kernel_launch(void* const* d_in, const int* in_sizes, int n_in,
                              void* d_out, int out_size, void* d_ws, size_t ws_size,
                              hipStream_t stream) {
  (void)in_sizes; (void)n_in; (void)out_size; (void)ws_size;
  const float* x      = (const float*)d_in[0];
  const float* Wih_e  = (const float*)d_in[1];
  const float* Whh_e  = (const float*)d_in[2];
  const float* bih_e  = (const float*)d_in[3];
  const float* bhh_e  = (const float*)d_in[4];
  const float* Wih_d  = (const float*)d_in[5];
  const float* Whh_d  = (const float*)d_in[6];
  const float* bih_d  = (const float*)d_in[7];
  const float* bhh_d  = (const float*)d_in[8];
  const float* WattnW = (const float*)d_in[9];
  const float* Wattnb = (const float*)d_in[10];
  const float* vvec   = (const float*)d_in[11];
  const float* WoutW  = (const float*)d_in[12];
  const float* Woutb  = (const float*)d_in[13];
  const float* WusW   = (const float*)d_in[14];
  const float* Wusb   = (const float*)d_in[15];
  float* out = (float*)d_out;

  // ---- workspace carve (256B aligned) ----
  size_t off = 0;
  char* base = (char*)d_ws;
  auto alloc = [&](size_t bytes) -> char* {
    char* p = base + off;
    off = (off + bytes + 255) & ~(size_t)255;
    return p;
  };
  _Float16* wih_e_h  = (_Float16*)alloc((size_t)G3 * FIN * 2);
  _Float16* whh_e_h  = (_Float16*)alloc((size_t)G3 * HID * 2);
  _Float16* wihd_c_h = (_Float16*)alloc((size_t)G3 * HID * 2);
  _Float16* whh_d_h  = (_Float16*)alloc((size_t)G3 * HID * 2);
  _Float16* w1_h     = (_Float16*)alloc((size_t)HID * HID * 2);
  _Float16* w2_h     = (_Float16*)alloc((size_t)HID * HID * 2);
  _Float16* wout_h   = (_Float16*)alloc((size_t)HID * 2 * HID * 2);
  float* wihd_p = (float*)alloc(G3 * 4);
  float* wout_p = (float*)alloc(HID * 4);
  float* hbuf0  = (float*)alloc((size_t)BSZ * HID * 4);
  float* hbuf1  = (float*)alloc((size_t)BSZ * HID * 4);
  float* encout = (float*)alloc((size_t)TSEQ * BSZ * HID * 4);
  float* eproj  = (float*)alloc((size_t)TSEQ * BSZ * HID * 4);
  float* hW1    = (float*)alloc((size_t)BSZ * HID * 4);
  float* score  = (float*)alloc((size_t)TSEQ * BSZ * 4);
  float* attn   = (float*)alloc((size_t)TSEQ * BSZ * 4);
  float* ctx    = (float*)alloc((size_t)BSZ * HID * 4);
  float* outv   = (float*)alloc((size_t)BSZ * HID * 4);
  float* pred   = (float*)alloc(BSZ * 4);
  float* hbufs[2] = {hbuf0, hbuf1};

  // ---- one-time weight repack to f16 ----
  k_cvt_f16<<<512, 256, 0, stream>>>(Wih_e, wih_e_h, G3, FIN, 0, FIN);
  k_cvt_f16<<<512, 256, 0, stream>>>(Whh_e, whh_e_h, G3, HID, 0, HID);
  k_cvt_f16<<<512, 256, 0, stream>>>(Wih_d, wihd_c_h, G3, 1 + HID, 1, HID);
  k_cvt_f16<<<512, 256, 0, stream>>>(Whh_d, whh_d_h, G3, HID, 0, HID);
  k_cvt_f16<<<256, 256, 0, stream>>>(WattnW, w1_h, HID, 2 * HID, 0, HID);
  k_cvt_f16<<<256, 256, 0, stream>>>(WattnW, w2_h, HID, 2 * HID, HID, HID);
  k_cvt_f16<<<512, 256, 0, stream>>>(WoutW, wout_h, HID, 1 + 2 * HID, 1, 2 * HID);
  k_gather_f32<<<4, 256, 0, stream>>>(Wih_d, wihd_p, G3, 1 + HID);
  k_gather_f32<<<1, 256, 0, stream>>>(WoutW, wout_p, HID, 1 + 2 * HID);

  hipMemsetAsync(hbuf0, 0, (size_t)BSZ * HID * 4, stream);

  // ---- encoder: 336 sequential fused GRU steps ----
  for (int t = 0; t < TSEQ; ++t) {
    k_gru_step<<<64, 256, 0, stream>>>(
        x + (size_t)t * FIN, TSEQ * FIN, FIN, wih_e_h,
        hbufs[t & 1], whh_e_h, bih_e, bhh_e,
        nullptr, nullptr,
        hbufs[(t + 1) & 1], encout + (size_t)t * BSZ * HID);
  }
  float* h_final = hbufs[0];  // 336 is even

  // ---- enc_proj = encoder_outputs @ W2^T + b  (172032 x 256 x 256) ----
  k_gemm<<<(TSEQ * BSZ / 16) * 2, 256, 0, stream>>>(
      encout, HID, w2_h, HID, Wattnb, eproj, HID);

  // ---- pred_usage = h_final @ Wusage^T + b ----
  k_pred<<<64, 256, 0, stream>>>(h_final, WusW, Wusb, pred, nullptr, 0);

  // ---- decoder: 48 sequential attention + GRU steps ----
  for (int s = 0; s < TL; ++s) {
    float* h_in  = hbufs[s & 1];
    float* h_out = hbufs[(s + 1) & 1];
    // hW1 = h @ W1^T
    k_gemm<<<64, 256, 0, stream>>>(h_in, HID, w1_h, HID, nullptr, hW1, HID);
    // score / softmax / context (HBM-bound streaming)
    k_score<<<TSEQ * BSZ / 8, 256, 0, stream>>>(hW1, eproj, vvec, score);
    k_softmax<<<64, 256, 0, stream>>>(score, attn);
    k_context<<<BSZ, 256, 0, stream>>>(attn, encout, ctx);
    // decoder GRU (context GEMM + rank-1 pred column)
    k_gru_step<<<64, 256, 0, stream>>>(
        ctx, HID, HID, wihd_c_h,
        h_in, whh_d_h, bih_d, bhh_d,
        pred, wihd_p, h_out, nullptr);
    // out = [pred, h, ctx] @ Wout^T + b
    k_dec_out<<<64, 256, 0, stream>>>(h_out, ctx, wout_h, wout_p, Woutb, pred, outv);
    // pred = out @ Wusage^T + b ; scatter into d_out[:, s]
    k_pred<<<64, 256, 0, stream>>>(outv, WusW, Wusb, pred, out + s, TL);
  }
}
